// lstm_generator_82471962018550
// MI455X (gfx1250) — matrix operand order repaired and checked by reading, not compiled, not run on hardware
//
#include <hip/hip_runtime.h>

// ---------------------------------------------------------------------------
// 2-layer LSTM (H=256, B=16, T=8192, D=7) persistent-kernel implementation
// for MI455X (gfx1250, wave32, WMMA).
//
// Strategy:
//  * 16 persistent workgroups, each owns 16 hidden dims (4 gate strips of 16
//    columns). Weights converted f32->f16 once and pinned in LDS in the exact
//    WMMA B-operand lane layout (zero weight traffic per step).
//  * Per step, per WG: layer0 GEMM  h(16x256) x W(256x64)  -> 8 wmma/wave,
//    layer1 GEMM [y0;h1](16x512) x W(512x64) -> 16 wmma/wave, f32 accum.
//  * h exchanged across WGs through double-buffered f16 buffers in L2;
//    grid barrier = atomic counter + acquire spin with s_sleep backoff.
//  * c-state lives in registers for the entire time loop.
//  * Head GEMM runs afterwards, parallel over all (t,b).
// ---------------------------------------------------------------------------

#define Hs   256
#define Bs   16
#define Ts   8192
#define Ds   7
#define NWG  16          // workgroups; each owns 16 hidden dims
#define KC0  8           // K=256 in chunks of 32 (layer 0)
#define KC1  16          // K=512 in chunks of 32 (layer 1, [Wih1;Whh1])
#define THREADS 128      // 4 waves (one per gate strip)
#define AROW 520         // padded f16 row stride for A staging (bank spread)

typedef __attribute__((ext_vector_type(16))) _Float16 v16h;
typedef __attribute__((ext_vector_type(8)))  _Float16 v8h;
typedef __attribute__((ext_vector_type(8)))  float    v8f;

__device__ __forceinline__ float sigf(float x){ return 1.0f / (1.0f + __expf(-x)); }

// Assemble a v16h WMMA operand from two contiguous 16B LDS blocks.
__device__ __forceinline__ v16h pack2(const _Float16* p0, const _Float16* p1){
  v8h a = *(const v8h*)p0;
  v8h b = *(const v8h*)p1;
  v16h r;
#pragma unroll
  for (int i = 0; i < 8; ++i){ r[i] = a[i]; r[i + 8] = b[i]; }
  return r;
}

__device__ __forceinline__ void gridbar(unsigned* cnt, unsigned gen){
  __threadfence();            // release all global stores (h broadcast)
  __syncthreads();
  if (threadIdx.x == 0){
    atomicAdd(cnt, 1u);
    while (__hip_atomic_load(cnt, __ATOMIC_ACQUIRE, __HIP_MEMORY_SCOPE_AGENT) < gen * NWG){
      __builtin_amdgcn_s_sleep(1);
    }
  }
  __syncthreads();
}

__global__ void lstm_init(_Float16* h0b, _Float16* h1b, unsigned* sync){
  int i = threadIdx.x;
  if (i == 0) *sync = 0u;
  for (int k = i; k < 2 * Bs * Hs; k += 256){
    h0b[k] = (_Float16)0.0f;
    h1b[k] = (_Float16)0.0f;
  }
}

__global__ __launch_bounds__(THREADS, 1)
void lstm_persist(const float* __restrict__ x,
                  const float* __restrict__ Wih0, const float* __restrict__ Whh0,
                  const float* __restrict__ b0,
                  const float* __restrict__ Wih1, const float* __restrict__ Whh1,
                  const float* __restrict__ b1,
                  float* __restrict__ out,
                  _Float16* __restrict__ h0b, _Float16* __restrict__ h1b,
                  _Float16* __restrict__ y1buf, unsigned* __restrict__ sync)
{
  // Weight slices in packed WMMA-B layout: [strip][kchunk][half][lane][8 f16]
  __shared__ __align__(16) _Float16 wl0[4 * KC0 * 512];   // 32 KB
  __shared__ __align__(16) _Float16 wl1[4 * KC1 * 512];   // 64 KB
  __shared__ __align__(16) _Float16 abuf[Bs * AROW];      // 16.25 KB (A operand)
  __shared__ float gbuf[4][Bs][17];                       // gate accum, bank-padded
  __shared__ float wih0s[4][16][8];                       // W_ih0 slice (7 + pad)
  __shared__ float bias0[4][16];
  __shared__ float bias1[4][16];
  __shared__ float xt[Bs][8];

  const int tid  = threadIdx.x;
  const int wg   = blockIdx.x;
  const int lane = tid & 31;
  const int wave = tid >> 5;      // 0..3 == gate strip (i,f,g,o)
  const int mlo  = lane & 15;     // M index (batch) / N index
  const int hi16 = lane >> 4;

  // ---- one-time staging: f32 -> f16, packed B-operand layout ----
  for (int idx = tid; idx < 4 * KC0 * 512; idx += THREADS){
    int e = idx & 7, ln = (idx >> 3) & 31, half = (idx >> 8) & 1;
    int kc = (idx >> 9) & (KC0 - 1), st = idx >> 12;
    int k = kc * 32 + ((ln < 16) ? 0 : 8) + half * 16 + e;
    int n = st * 256 + wg * 16 + (ln & 15);
    wl0[idx] = (_Float16)Whh0[n * Hs + k];
  }
  for (int idx = tid; idx < 4 * KC1 * 512; idx += THREADS){
    int e = idx & 7, ln = (idx >> 3) & 31, half = (idx >> 8) & 1;
    int kc = (idx >> 9) & (KC1 - 1), st = idx >> 13;
    int k = kc * 32 + ((ln < 16) ? 0 : 8) + half * 16 + e;   // k in [0,512)
    int n = st * 256 + wg * 16 + (ln & 15);
    wl1[idx] = (_Float16)((k < Hs) ? Wih1[n * Hs + k] : Whh1[n * Hs + (k - Hs)]);
  }
  for (int idx = tid; idx < 4 * 16; idx += THREADS){
    int st = idx >> 4, j = idx & 15;
    int n = st * 256 + wg * 16 + j;
    bias0[st][j] = b0[n];
    bias1[st][j] = b1[n];
#pragma unroll
    for (int d = 0; d < 8; ++d) wih0s[st][j][d] = (d < Ds) ? Wih0[n * Ds + d] : 0.0f;
  }
  __syncthreads();

  float c0r[2] = {0.0f, 0.0f};
  float c1r[2] = {0.0f, 0.0f};
  unsigned gen = 0;

  for (int t = 0; t < Ts; ++t){
    const int par = t & 1;

    // ================= layer 0 =================
    if (tid < Bs * Ds){
      int bb = tid / Ds, d = tid % Ds;
      xt[bb][d] = x[(bb * Ts + t) * Ds + d];
    }
    { // stage h0 (prev) into padded A buffer
      const _Float16* hsrc = h0b + par * Bs * Hs;
      for (int idx = tid; idx < Bs * Hs / 8; idx += THREADS){
        int m = idx >> 5, c8 = (idx & 31) * 8;
        *(v8h*)(abuf + m * AROW + c8) = *(const v8h*)(hsrc + m * Hs + c8);
      }
    }
    __syncthreads();
    {
      v8f acc = {};
#pragma unroll
      for (int kc = 0; kc < KC0; ++kc){
        int kb = kc * 32 + hi16 * 8;
        v16h A  = pack2(abuf + mlo * AROW + kb, abuf + mlo * AROW + kb + 16);
        const _Float16* wp = wl0 + (wave * KC0 + kc) * 512 + lane * 8;
        v16h Bf = pack2(wp, wp + 256);
        acc = __builtin_amdgcn_wmma_f32_16x16x32_f16(false, A, false, Bf,
                                                     (short)0, acc, false, false);
      }
#pragma unroll
      for (int v = 0; v < 8; ++v) gbuf[wave][hi16 * 8 + v][mlo] = acc[v];
    }
    __syncthreads();
#pragma unroll
    for (int pp = 0; pp < 2; ++pp){
      int p  = tid + pp * THREADS;
      int bb = p >> 4, j = p & 15;
      float g[4];
#pragma unroll
      for (int s = 0; s < 4; ++s){
        float a = gbuf[s][bb][j] + bias0[s][j];
#pragma unroll
        for (int d = 0; d < Ds; ++d) a += xt[bb][d] * wih0s[s][j][d];
        g[s] = a;
      }
      float cc = sigf(g[1]) * c0r[pp] + sigf(g[0]) * tanhf(g[2]);
      c0r[pp] = cc;
      float hh = sigf(g[3]) * tanhf(cc);
      h0b[(par ^ 1) * Bs * Hs + bb * Hs + wg * 16 + j] = (_Float16)hh;
      if (t == Ts - 1){
        out[917504 + 0 * Bs * Hs + bb * Hs + wg * 16 + j] = hh;          // h_n[0]
        out[917504 + 8192 + 0 * Bs * Hs + bb * Hs + wg * 16 + j] = cc;   // c_n[0]
      }
    }
    gridbar(sync, ++gen);

    // ================= layer 1 =================
    { // stage A = [ y0_t (new h0) | h1 prev ], K = 512
      const _Float16* y0 = h0b + (par ^ 1) * Bs * Hs;
      const _Float16* h1 = h1b + par * Bs * Hs;
      for (int idx = tid; idx < Bs * 512 / 8; idx += THREADS){
        int m = idx >> 6, c8 = (idx & 63) * 8;
        const _Float16* src = (c8 < Hs) ? (y0 + m * Hs + c8) : (h1 + m * Hs + (c8 - Hs));
        *(v8h*)(abuf + m * AROW + c8) = *(const v8h*)src;
      }
    }
    __syncthreads();
    {
      v8f acc = {};
#pragma unroll
      for (int kc = 0; kc < KC1; ++kc){
        int kb = kc * 32 + hi16 * 8;
        v16h A  = pack2(abuf + mlo * AROW + kb, abuf + mlo * AROW + kb + 16);
        const _Float16* wp = wl1 + (wave * KC1 + kc) * 512 + lane * 8;
        v16h Bf = pack2(wp, wp + 256);
        acc = __builtin_amdgcn_wmma_f32_16x16x32_f16(false, A, false, Bf,
                                                     (short)0, acc, false, false);
      }
#pragma unroll
      for (int v = 0; v < 8; ++v) gbuf[wave][hi16 * 8 + v][mlo] = acc[v];
    }
    __syncthreads();
#pragma unroll
    for (int pp = 0; pp < 2; ++pp){
      int p  = tid + pp * THREADS;
      int bb = p >> 4, j = p & 15;
      float g[4];
#pragma unroll
      for (int s = 0; s < 4; ++s) g[s] = gbuf[s][bb][j] + bias1[s][j];
      float cc = sigf(g[1]) * c1r[pp] + sigf(g[0]) * tanhf(g[2]);
      c1r[pp] = cc;
      float hh = sigf(g[3]) * tanhf(cc);
      h1b[(par ^ 1) * Bs * Hs + bb * Hs + wg * 16 + j] = (_Float16)hh;
      y1buf[(t * Bs + bb) * Hs + wg * 16 + j] = (_Float16)hh;
      if (t == Ts - 1){
        out[917504 + 1 * Bs * Hs + bb * Hs + wg * 16 + j] = hh;          // h_n[1]
        out[917504 + 8192 + 1 * Bs * Hs + bb * Hs + wg * 16 + j] = cc;   // c_n[1]
      }
    }
    gridbar(sync, ++gen);
  }
}

__global__ __launch_bounds__(256)
void lstm_head(const _Float16* __restrict__ y1buf, const float* __restrict__ Whead,
               const float* __restrict__ bhead, float* __restrict__ out)
{
  __shared__ float wsm[Ds][Hs];
  __shared__ float bsm[Ds];
  for (int idx = threadIdx.x; idx < Ds * Hs; idx += 256) wsm[idx >> 8][idx & 255] = Whead[idx];
  if (threadIdx.x < Ds) bsm[threadIdx.x] = bhead[threadIdx.x];
  __syncthreads();

  int p  = blockIdx.x * 256 + threadIdx.x;   // p = t*16 + b
  int t  = p >> 4, bb = p & 15;
  const _Float16* y = y1buf + (size_t)p * Hs;
  float acc[Ds];
#pragma unroll
  for (int o = 0; o < Ds; ++o) acc[o] = bsm[o];
  for (int k = 0; k < Hs; k += 8){
    v8h yv = *(const v8h*)(y + k);
#pragma unroll
    for (int e = 0; e < 8; ++e){
      float v = (float)yv[e];
#pragma unroll
      for (int o = 0; o < Ds; ++o) acc[o] += v * wsm[o][k + e];
    }
  }
#pragma unroll
  for (int o = 0; o < Ds; ++o) out[((size_t)bb * Ts + t) * Ds + o] = acc[o];
}

extern "C" void kernel_launch(void* const* d_in, const int* in_sizes, int n_in,
                              void* d_out, int out_size, void* d_ws, size_t ws_size,
                              hipStream_t stream)
{
  (void)in_sizes; (void)n_in; (void)out_size; (void)ws_size;
  const float* x    = (const float*)d_in[0];
  const float* Wih0 = (const float*)d_in[1];
  const float* Whh0 = (const float*)d_in[2];
  const float* b0   = (const float*)d_in[3];
  const float* Wih1 = (const float*)d_in[4];
  const float* Whh1 = (const float*)d_in[5];
  const float* b1   = (const float*)d_in[6];
  const float* Wh   = (const float*)d_in[7];
  const float* bh   = (const float*)d_in[8];
  float* out = (float*)d_out;

  char* ws = (char*)d_ws;
  unsigned*  sync = (unsigned*)ws;                                   // barrier counter
  _Float16*  h0b  = (_Float16*)(ws + 256);                           // 2*B*H f16
  _Float16*  h1b  = (_Float16*)(ws + 256 + 2 * Bs * Hs * 2);         // 2*B*H f16
  _Float16*  y1   = (_Float16*)(ws + 256 + 4 * Bs * Hs * 2 + 256);   // T*B*H f16 (64 MB)

  hipLaunchKernelGGL(lstm_init, dim3(1), dim3(256), 0, stream, h0b, h1b, sync);
  hipLaunchKernelGGL(lstm_persist, dim3(NWG), dim3(THREADS), 0, stream,
                     x, Wih0, Whh0, b0, Wih1, Whh1, b1, out, h0b, h1b, y1, sync);
  hipLaunchKernelGGL(lstm_head, dim3((Ts * Bs) / 256), dim3(256), 0, stream,
                     y1, Wh, bh, out);
}
